// DynamicDepthSeparableConv1dMultiheadAttention_9414568313152
// MI455X (gfx1250) — compile-verified, matrix operand order checked by reading, not run on hardware
//
#include <hip/hip_runtime.h>
#include <hip/hip_bf16.h>

// ---------------------------------------------------------------------------
// CDNA5 (gfx1250) implementation:
//   K1 proj_gemm : P[proj,b,o,l] = pw[o,c] * x[b,c,l]            (bf16 WMMA)
//   K2 dwconv    : gated depthwise conv 3/15 + scale + relayout  (VALU)
//   K3 attn      : flash attention, double-buffered async LDS    (bf16 WMMA)
//   K4 unify     : out[b,c,l] = w_u[c,o] * O[b,l,o] + b_u[c]     (bf16 WMMA)
// ---------------------------------------------------------------------------

typedef __bf16 bf16_t;
typedef bf16_t v16bf __attribute__((ext_vector_type(16)));
typedef bf16_t v8bf  __attribute__((ext_vector_type(8)));
typedef float  v8f   __attribute__((ext_vector_type(8)));

#define B_   16
#define C_   128
#define H_   8
#define CO_  1024
#define L_   1024
#define BH_  128

#ifndef __has_builtin
#define __has_builtin(x) 0
#endif
#if __has_builtin(__builtin_amdgcn_global_load_async_to_lds_b128)
#define ASYNC_LDS 1
#else
#define ASYNC_LDS 0
#endif
#if __has_builtin(__builtin_amdgcn_s_wait_asynccnt)
#define HAVE_WAIT_ASYNC_BUILTIN 1
#else
#define HAVE_WAIT_ASYNC_BUILTIN 0
#endif

// 16-byte global -> LDS copy; async DMA (ASYNCcnt) when available.
// Builtin signature (from clang diagnostic): param 1 is a pointer to a
// 16-byte int vector in the global (AS1) address space; param 2 the LDS
// (AS3) counterpart; then imm offset + imm cpol.
static __device__ __forceinline__ void async_cp16(bf16_t* lds_dst,
                                                  const bf16_t* gsrc) {
#if ASYNC_LDS
  typedef int v4i __attribute__((vector_size(16)));
  __builtin_amdgcn_global_load_async_to_lds_b128(
      (__attribute__((address_space(1))) v4i*)gsrc,
      (__attribute__((address_space(3))) v4i*)lds_dst, 0, 0);
#else
  *(float4*)lds_dst = *(const float4*)gsrc;
#endif
}

static __device__ __forceinline__ void wait_async() {
#if ASYNC_LDS
#if HAVE_WAIT_ASYNC_BUILTIN
  __builtin_amdgcn_s_wait_asynccnt(0);
#else
  asm volatile("s_wait_asynccnt 0x0" ::: "memory");
#endif
#endif
}

static __device__ __forceinline__ v8f zero8() {
  v8f z = {0.f, 0.f, 0.f, 0.f, 0.f, 0.f, 0.f, 0.f};
  return z;
}

// D = A*B + C, 16x16x32 bf16 -> f32
static __device__ __forceinline__ v8f wmma_bf16(v16bf a, v16bf b, v8f c) {
  return __builtin_amdgcn_wmma_f32_16x16x32_bf16(false, a, false, b,
                                                 (short)0, c, false, false);
}

// Fragment loader for 16-bit A(16x32) / B(32x16) operands.
// Lanes 0-15 take K{0..7,16..23}, lanes 16-31 K{8..15,24..31};
// `row` points at the 32-element K-chunk for this lane's M/N index.
static __device__ __forceinline__ v16bf frag32(const bf16_t* row, int lane) {
  const bf16_t* p = row + ((lane & 16) ? 8 : 0);
  v8bf lo = *(const v8bf*)(p);
  v8bf hi = *(const v8bf*)(p + 16);
  v16bf f;
#pragma unroll
  for (int i = 0; i < 8; ++i) { f[i] = lo[i]; f[8 + i] = hi[i]; }
  return f;
}

// ---------------------------------------------------------------------------
// Kernel 1: pointwise projection GEMM.  grid = (CO/128, L/128, 3*B)
// P[proj][b][o][l] = sum_c pw[o][c] * x[b][c][l]   (bf16 out)
// ---------------------------------------------------------------------------
__global__ __launch_bounds__(256) void proj_gemm_kernel(
    const float* __restrict__ x, const float* __restrict__ pw0,
    const float* __restrict__ pw1, const float* __restrict__ pw2,
    bf16_t* __restrict__ pbuf) {
  __shared__ alignas(16) bf16_t a_lds[128 * 32];
  __shared__ alignas(16) bf16_t b_lds[128 * 32];

  const int tid = threadIdx.x;
  const int lane = tid & 31, wave = tid >> 5;
  const int mBlock = blockIdx.x * 128;
  const int lBlock = blockIdx.y * 128;
  const int z = blockIdx.z;
  const int proj = z >> 4;
  const int b = z & 15;
  const float* pw = (proj == 0) ? pw0 : (proj == 1) ? pw1 : pw2;

  const int waveM = (wave >> 1) * 32;
  const int waveN = (wave & 1) * 64;

  v8f acc[2][4];
#pragma unroll
  for (int mt = 0; mt < 2; ++mt)
#pragma unroll
    for (int nt = 0; nt < 4; ++nt) acc[mt][nt] = zero8();

  for (int kk0 = 0; kk0 < C_; kk0 += 32) {
    {  // stage A tile: pw[mBlock..+128][kk0..+32] -> a_lds[m][kk]
      int row = tid >> 1;
      int cb = (tid & 1) * 16;
      const float* src = pw + (size_t)(mBlock + row) * C_ + kk0 + cb;
#pragma unroll
      for (int i = 0; i < 16; ++i) a_lds[row * 32 + cb + i] = (bf16_t)src[i];
    }
    {  // stage B tile transposed: x[kk0..+32][lBlock..+128] -> b_lds[l][kk]
      int c = tid >> 3;
      int lb = (tid & 7) * 16;
      const float* src = x + ((size_t)b * C_ + kk0 + c) * L_ + lBlock + lb;
#pragma unroll
      for (int i = 0; i < 16; ++i) b_lds[(lb + i) * 32 + c] = (bf16_t)src[i];
      if (kk0 + 32 < C_) __builtin_prefetch(src + 32 * L_, 0, 0);
    }
    __syncthreads();

    v16bf af[2], bfrag[4];
#pragma unroll
    for (int mt = 0; mt < 2; ++mt)
      af[mt] = frag32(&a_lds[(waveM + mt * 16 + (lane & 15)) * 32], lane);
#pragma unroll
    for (int nt = 0; nt < 4; ++nt)
      bfrag[nt] = frag32(&b_lds[(waveN + nt * 16 + (lane & 15)) * 32], lane);
#pragma unroll
    for (int mt = 0; mt < 2; ++mt)
#pragma unroll
      for (int nt = 0; nt < 4; ++nt)
        acc[mt][nt] = wmma_bf16(af[mt], bfrag[nt], acc[mt][nt]);
    __syncthreads();
  }

  bf16_t* dst = pbuf + (size_t)(proj * B_ + b) * CO_ * L_;
  const int hi8 = (lane & 16) ? 8 : 0;
#pragma unroll
  for (int mt = 0; mt < 2; ++mt)
#pragma unroll
    for (int nt = 0; nt < 4; ++nt)
#pragma unroll
      for (int r = 0; r < 8; ++r) {
        int o = mBlock + waveM + mt * 16 + r + hi8;
        int l = lBlock + waveN + nt * 16 + (lane & 15);
        dst[(size_t)o * L_ + l] = (bf16_t)acc[mt][nt][r];
      }
}

// ---------------------------------------------------------------------------
// Kernel 2: gated depthwise conv (k=3, k=15) + scaling + relayout.
// grid = (L/256, CO, 3*B)
//   proj 0 -> Qt[bh][l][c] * c^-0.25
//   proj 1 -> Kt[bh][l][c] * c^-0.25
//   proj 2 -> V [bh][c][l]
// ---------------------------------------------------------------------------
__global__ __launch_bounds__(256) void dwconv_kernel(
    const bf16_t* __restrict__ pbuf,
    const float* __restrict__ d3q, const float* __restrict__ d15q, const float* __restrict__ gq,
    const float* __restrict__ d3k, const float* __restrict__ d15k, const float* __restrict__ gk,
    const float* __restrict__ d3v, const float* __restrict__ d15v, const float* __restrict__ gv,
    bf16_t* __restrict__ Qt, bf16_t* __restrict__ Kt, bf16_t* __restrict__ Vb) {
  const int l = blockIdx.x * 256 + threadIdx.x;
  const int o = blockIdx.y;
  const int z = blockIdx.z;
  const int proj = z >> 4;
  const int b = z & 15;

  const float* d3 = (proj == 0) ? d3q : (proj == 1) ? d3k : d3v;
  const float* d15 = (proj == 0) ? d15q : (proj == 1) ? d15k : d15v;
  const float* g = (proj == 0) ? gq : (proj == 1) ? gk : gv;

  float ga = g[0], gb = g[1];
  float mx = fmaxf(ga, gb);
  float ea = __expf(ga - mx), eb = __expf(gb - mx);
  float inv = 1.0f / (ea + eb);
  float g0 = ea * inv, g1 = eb * inv;

  const bf16_t* src = pbuf + ((size_t)(proj * B_ + b) * CO_ + o) * L_;

  float a3 = 0.f;
#pragma unroll
  for (int j = 0; j < 3; ++j) {
    int idx = l + j - 1;
    if ((unsigned)idx < (unsigned)L_) a3 += d3[o * 3 + j] * (float)src[idx];
  }
  float a15 = 0.f;
#pragma unroll
  for (int j = 0; j < 15; ++j) {
    int idx = l + j - 7;
    if ((unsigned)idx < (unsigned)L_) a15 += d15[o * 15 + j] * (float)src[idx];
  }
  float y = g0 * a3 + g1 * a15;

  const int cc = o & 127;
  const int h = o >> 7;
  const int bh = b * H_ + h;
  const float kInvScale = 0.29730177875068026f;  // 128^-0.25

  if (proj == 0)
    Qt[((size_t)bh * L_ + l) * C_ + cc] = (bf16_t)(y * kInvScale);
  else if (proj == 1)
    Kt[((size_t)bh * L_ + l) * C_ + cc] = (bf16_t)(y * kInvScale);
  else
    Vb[((size_t)bh * C_ + cc) * L_ + l] = (bf16_t)y;
}

// ---------------------------------------------------------------------------
// Kernel 3: flash attention.  grid = (L/128, BH), 8 waves x 16 q-cols each.
// Double-buffered K/V tiles staged with async global->LDS DMA.
// Output stored transposed: ObT[b][l][o]  (o = h*128 + c), bf16.
// ---------------------------------------------------------------------------
__global__ __launch_bounds__(256) void attn_kernel(
    const bf16_t* __restrict__ Qt, const bf16_t* __restrict__ Kt,
    const bf16_t* __restrict__ Vb, bf16_t* __restrict__ ObT) {
  __shared__ alignas(16) bf16_t klds[2][32 * 128];  // [buf][k][c]
  __shared__ alignas(16) bf16_t vlds[2][128 * 32];  // [buf][c][v]

  const int tid = threadIdx.x;
  const int lane = tid & 31, wave = tid >> 5;
  const int bh = blockIdx.y;
  const int qBase = blockIdx.x * 128 + wave * 16;

  auto stage = [&](int buf, int kb) {
    int row = tid >> 3, ch = (tid & 7) * 16;
    const bf16_t* ks = Kt + ((size_t)bh * L_ + kb + row) * C_ + ch;
    async_cp16(&klds[buf][row * 128 + ch], ks);
    async_cp16(&klds[buf][row * 128 + ch + 8], ks + 8);
    int c = tid >> 1, vh = (tid & 1) * 16;
    const bf16_t* vs = Vb + ((size_t)bh * C_ + c) * L_ + kb + vh;
    async_cp16(&vlds[buf][c * 32 + vh], vs);
    async_cp16(&vlds[buf][c * 32 + vh + 8], vs + 8);
  };

  // Q operand fragments (B layout), resident for the whole kernel.
  const bf16_t* qrow = Qt + ((size_t)bh * L_ + qBase + (lane & 15)) * C_;
  v16bf qf[4];
#pragma unroll
  for (int kc = 0; kc < 4; ++kc) qf[kc] = frag32(qrow + kc * 32, lane);

  v8f oacc[8];
#pragma unroll
  for (int t = 0; t < 8; ++t) oacc[t] = zero8();
  float m_run = -1e30f, l_run = 0.f;

  stage(0, 0);
  int buf = 0;
  for (int kb = 0; kb < L_; kb += 32, buf ^= 1) {
    wait_async();
    __syncthreads();  // current buffer ready; previous buffer fully consumed
    if (kb + 32 < L_) stage(buf ^ 1, kb + 32);

    // S[k,q]: two 16x16 tiles over 32 keys, reduction over c = 128.
    v8f s0 = zero8(), s1 = zero8();
#pragma unroll
    for (int kc = 0; kc < 4; ++kc) {
      v16bf a0 = frag32(&klds[buf][(lane & 15) * 128 + kc * 32], lane);
      v16bf a1 = frag32(&klds[buf][(16 + (lane & 15)) * 128 + kc * 32], lane);
      s0 = wmma_bf16(a0, qf[kc], s0);
      s1 = wmma_bf16(a1, qf[kc], s1);
    }

    // online softmax over the key axis for this lane's q column.
    float mloc = -1e30f;
#pragma unroll
    for (int r = 0; r < 8; ++r) mloc = fmaxf(mloc, fmaxf(s0[r], s1[r]));
    mloc = fmaxf(mloc, __shfl_xor(mloc, 16, 32));
    float newM = fmaxf(m_run, mloc);
    float alpha = __expf(m_run - newM);
    float ssum = 0.f;
#pragma unroll
    for (int r = 0; r < 8; ++r) {
      s0[r] = __expf(s0[r] - newM);
      s1[r] = __expf(s1[r] - newM);
      ssum += s0[r] + s1[r];
    }
    ssum += __shfl_xor(ssum, 16, 32);
    l_run = l_run * alpha + ssum;
    m_run = newM;
#pragma unroll
    for (int t = 0; t < 8; ++t)
#pragma unroll
      for (int r = 0; r < 8; ++r) oacc[t][r] *= alpha;

    // P (f32, D layout) -> B operand (bf16): lane-local packing.
    v16bf pf;
#pragma unroll
    for (int r = 0; r < 8; ++r) {
      pf[r] = (bf16_t)s0[r];
      pf[8 + r] = (bf16_t)s1[r];
    }

    // O[c,q] += V[c, kb..kb+32] * P
#pragma unroll
    for (int ct = 0; ct < 8; ++ct) {
      v16bf va = frag32(&vlds[buf][(ct * 16 + (lane & 15)) * 32], lane);
      oacc[ct] = wmma_bf16(va, pf, oacc[ct]);
    }
  }

  // Epilogue: normalize and store transposed [b][l][o]; consecutive accum
  // rows are consecutive o channels -> one 16B store per tile per lane.
  const float invl = 1.0f / l_run;
  const int hi8 = (lane & 16) ? 8 : 0;
  const int qcol = qBase + (lane & 15);
  const int b = bh >> 3, h = bh & 7;
  bf16_t* dst = ObT + ((size_t)b * L_ + qcol) * CO_ + (size_t)h * C_;
#pragma unroll
  for (int ct = 0; ct < 8; ++ct) {
    v8bf pk;
#pragma unroll
    for (int r = 0; r < 8; ++r) pk[r] = (bf16_t)(oacc[ct][r] * invl);
    *(v8bf*)(dst + ct * 16 + hi8) = pk;
  }
}

// ---------------------------------------------------------------------------
// Kernel 4: unify heads.  grid = (L/128, B).  M=C=128, K=CO=1024.
// out[b][c][l] = sum_o w_u[c][o] * ObT[b][l][o] + b_u[c]
// B tiles are K-contiguous in ObT -> async copies, no transpose.
// ---------------------------------------------------------------------------
__global__ __launch_bounds__(256) void unify_kernel(
    const bf16_t* __restrict__ ObT, const float* __restrict__ wu,
    const float* __restrict__ bu, float* __restrict__ out) {
  __shared__ alignas(16) bf16_t a_lds[128 * 32];
  __shared__ alignas(16) bf16_t b_lds[128 * 32];

  const int tid = threadIdx.x;
  const int lane = tid & 31, wave = tid >> 5;
  const int lBlock = blockIdx.x * 128;
  const int b = blockIdx.y;
  const int waveM = (wave >> 1) * 32;
  const int waveN = (wave & 1) * 64;

  v8f acc[2][4];
#pragma unroll
  for (int mt = 0; mt < 2; ++mt)
#pragma unroll
    for (int nt = 0; nt < 4; ++nt) acc[mt][nt] = zero8();

  for (int kk0 = 0; kk0 < CO_; kk0 += 32) {
    {  // B tile (async): ObT[b][lBlock+lrow][kk0..+32] -> b_lds[l][kk]
      int lrow = tid >> 1;
      int ob = (tid & 1) * 16;
      const bf16_t* src =
          ObT + ((size_t)b * L_ + lBlock + lrow) * CO_ + kk0 + ob;
      async_cp16(&b_lds[lrow * 32 + ob], src);
      async_cp16(&b_lds[lrow * 32 + ob + 8], src + 8);
    }
    {  // A tile (sync, converts f32->bf16): w_u[row][kk0..+32]
      int row = tid >> 1;
      int cb = (tid & 1) * 16;
      const float* src = wu + (size_t)row * CO_ + kk0 + cb;
#pragma unroll
      for (int i = 0; i < 16; ++i) a_lds[row * 32 + cb + i] = (bf16_t)src[i];
    }
    wait_async();
    __syncthreads();

    v16bf af[2], bfrag[4];
#pragma unroll
    for (int mt = 0; mt < 2; ++mt)
      af[mt] = frag32(&a_lds[(waveM + mt * 16 + (lane & 15)) * 32], lane);
#pragma unroll
    for (int nt = 0; nt < 4; ++nt)
      bfrag[nt] = frag32(&b_lds[(waveN + nt * 16 + (lane & 15)) * 32], lane);
#pragma unroll
    for (int mt = 0; mt < 2; ++mt)
#pragma unroll
      for (int nt = 0; nt < 4; ++nt)
        acc[mt][nt] = wmma_bf16(af[mt], bfrag[nt], acc[mt][nt]);
    __syncthreads();
  }

  const int hi8 = (lane & 16) ? 8 : 0;
#pragma unroll
  for (int mt = 0; mt < 2; ++mt)
#pragma unroll
    for (int nt = 0; nt < 4; ++nt)
#pragma unroll
      for (int r = 0; r < 8; ++r) {
        int c = waveM + mt * 16 + r + hi8;
        int l = lBlock + waveN + nt * 16 + (lane & 15);
        out[((size_t)b * C_ + c) * L_ + l] = acc[mt][nt][r] + bu[c];
      }
}

// ---------------------------------------------------------------------------
extern "C" void kernel_launch(void* const* d_in, const int* in_sizes, int n_in,
                              void* d_out, int out_size, void* d_ws,
                              size_t ws_size, hipStream_t stream) {
  (void)in_sizes; (void)n_in; (void)out_size; (void)ws_size;

  const float* x    = (const float*)d_in[0];
  const float* pw_q = (const float*)d_in[1];
  const float* d3q  = (const float*)d_in[2];
  const float* d15q = (const float*)d_in[3];
  const float* gq   = (const float*)d_in[4];
  const float* pw_k = (const float*)d_in[5];
  const float* d3k  = (const float*)d_in[6];
  const float* d15k = (const float*)d_in[7];
  const float* gk   = (const float*)d_in[8];
  const float* pw_v = (const float*)d_in[9];
  const float* d3v  = (const float*)d_in[10];
  const float* d15v = (const float*)d_in[11];
  const float* gv   = (const float*)d_in[12];
  const float* wu   = (const float*)d_in[13];
  const float* bu   = (const float*)d_in[14];

  char* ws = (char*)d_ws;
  const size_t PBYTES = (size_t)3 * B_ * CO_ * L_ * sizeof(bf16_t);  // 96 MB
  const size_t TBYTES = (size_t)BH_ * L_ * C_ * sizeof(bf16_t);      // 32 MB
  bf16_t* pbuf = (bf16_t*)ws;
  bf16_t* Qt   = (bf16_t*)(ws + PBYTES);
  bf16_t* Kt   = (bf16_t*)(ws + PBYTES + TBYTES);
  bf16_t* Vb   = (bf16_t*)(ws + PBYTES + 2 * TBYTES);
  bf16_t* ObT  = (bf16_t*)ws;  // alias: pbuf is dead after dwconv_kernel

  proj_gemm_kernel<<<dim3(CO_ / 128, L_ / 128, 3 * B_), 256, 0, stream>>>(
      x, pw_q, pw_k, pw_v, pbuf);
  dwconv_kernel<<<dim3(L_ / 256, CO_, 3 * B_), 256, 0, stream>>>(
      pbuf, d3q, d15q, gq, d3k, d15k, gk, d3v, d15v, gv, Qt, Kt, Vb);
  attn_kernel<<<dim3(L_ / 128, BH_, 1), 256, 0, stream>>>(Qt, Kt, Vb, ObT);
  unify_kernel<<<dim3(L_ / 128, B_, 1), 256, 0, stream>>>(ObT, wu, bu,
                                                          (float*)d_out);
}
